// MultiModalClinicalGCN_67757404062364
// MI455X (gfx1250) — matrix operand order repaired and checked by reading, not compiled, not run on hardware
//
#include <hip/hip_runtime.h>
#include <hip/hip_bf16.h>
#include <stdint.h>

typedef __attribute__((ext_vector_type(16))) _Float16 v16h;
typedef __attribute__((ext_vector_type(8)))  _Float16 v8h;
typedef __attribute__((ext_vector_type(4)))  _Float16 v4h;
typedef __attribute__((ext_vector_type(8)))  float    v8f;

#define LDS_ADDR32(p) \
  ((unsigned)(uintptr_t)(__attribute__((address_space(3))) const void*)(p))

// ---------------------------------------------------------------------------
// Weight pre-swizzle: f32 row-major [K, NW] -> f16 in WMMA B-fragment order.
// Fragment element j of lane L, wave-tile wt, kstep ks corresponds to
//   K = ks*32 + 16*(L/16) + j,   N = 16*wt + (L%16)
// Stored so each lane reads its 16 halves (32B) contiguously; per-kstep
// stride is 8*32*16 halves = 8192 bytes (fits IOFFSET immediates).
// ---------------------------------------------------------------------------
__global__ void swizzle_w_kernel(const float* __restrict__ W,
                                 _Float16* __restrict__ Wh, int K, int NW) {
  int idx = blockIdx.x * blockDim.x + threadIdx.x;
  if (idx >= K * NW) return;
  int k = idx / NW, ncol = idx % NW;
  int ks = k >> 5, kl = k & 31;
  int lane = ((kl >> 4) << 4) | (ncol & 15);
  int j = kl & 15;
  int wt = ncol >> 4;
  int NT = NW >> 4;
  Wh[((size_t)((ks * NT + wt) * 32 + lane) << 4) + j] = (_Float16)W[idx];
}

__global__ void fill_f32_kernel(float* __restrict__ p, float v, int n) {
  int i = blockIdx.x * blockDim.x + threadIdx.x;
  if (i < n) p[i] = v;
}

__global__ void deg_count_kernel(const long long* __restrict__ dst,
                                 float* __restrict__ deg, int e) {
  int i = blockIdx.x * blockDim.x + threadIdx.x;
  if (i < e) atomicAdd(&deg[dst[i]], 1.0f);
}

__global__ void rsqrt_kernel(float* __restrict__ p, int n) {
  int i = blockIdx.x * blockDim.x + threadIdx.x;
  if (i < n) p[i] = rsqrtf(p[i]);
}

// ---------------------------------------------------------------------------
// Generic WMMA GEMM: out[m, 0:128] = act(A[m, 0:K] @ Wsw + bias)
// A columns [0, SPLIT) from A32 (f32, width a32w), [SPLIT, K) from A16 (f16,
// width a16w). SPLIT is a multiple of 64 so each 64-wide K stage is uniform.
// Block = 256 threads (8 waves) -> 32 M-rows x 128 N-cols. Wave wv owns
// N-tile [16wv,16wv+16) and two 16-row A sub-tiles (B fragment reused twice).
// f32 A stages: load+cvt+ds_store (+ prefetch of next stage).
// f16 A stages: GLOBAL_LOAD_ASYNC_TO_LDS_B128 (ASYNCcnt) per lane.
// Optional fused GCN self-loop init: accout[m,c] = dinv[m]^2 * v + bias2[c].
// ---------------------------------------------------------------------------
template <int K, int SPLIT>
__global__ void __launch_bounds__(256)
gemm_wmma_kernel(const float* __restrict__ A32, int a32w,
                 const _Float16* __restrict__ A16, int a16w,
                 const _Float16* __restrict__ Wsw,
                 const float* __restrict__ bias,
                 float* __restrict__ out32, _Float16* __restrict__ out16,
                 int relu,
                 float* __restrict__ accout, const float* __restrict__ dinv,
                 const float* __restrict__ bias2) {
  __shared__ _Float16 lds[32 * 72];  // 32 rows x 64 halves, pad stride 72
  const int tid  = threadIdx.x;
  const int wv   = tid >> 5;        // wave 0..7 -> N-tile
  const int lane = tid & 31;
  const int g    = lane >> 4;       // lane group 0/1
  const int ln   = lane & 15;
  const int m0   = blockIdx.x << 5;
  // cooperative A loader: 32 rows x 64 halves, 8 halves per thread
  const int lr = tid >> 3;          // row 0..31
  const int lc = (tid & 7) << 3;    // col 0,8,...,56

  const _Float16* bbase = Wsw + ((size_t)((wv << 5) + lane) << 4);
  const unsigned ldsDst = LDS_ADDR32(&lds[lr * 72 + lc]);

  v8f c0 = {0.f, 0.f, 0.f, 0.f, 0.f, 0.f, 0.f, 0.f};
  v8f c1 = {0.f, 0.f, 0.f, 0.f, 0.f, 0.f, 0.f, 0.f};

  for (int kb = 0; kb < K; kb += 64) {
    __syncthreads();
    if (kb < SPLIT) {  // f32 source stage (uniform)
      const float* ap = A32 + (size_t)(m0 + lr) * a32w + kb + lc;
      const float4 f0 = *(const float4*)ap;
      const float4 f1 = *(const float4*)(ap + 4);
      if (kb + 64 < SPLIT) __builtin_prefetch(ap + 64, 0, 1);
      v8h hv;
      hv[0] = (_Float16)f0.x; hv[1] = (_Float16)f0.y;
      hv[2] = (_Float16)f0.z; hv[3] = (_Float16)f0.w;
      hv[4] = (_Float16)f1.x; hv[5] = (_Float16)f1.y;
      hv[6] = (_Float16)f1.z; hv[7] = (_Float16)f1.w;
      *(v8h*)(&lds[lr * 72 + lc]) = hv;
    } else {           // f16 source stage: async copy straight into LDS
      const _Float16* ap =
          A16 + (size_t)(m0 + lr) * a16w + (kb - SPLIT) + lc;
      asm volatile("global_load_async_to_lds_b128 %0, %1, off"
                   :: "v"(ldsDst), "v"((unsigned long long)(uintptr_t)ap)
                   : "memory");
      asm volatile("s_wait_asynccnt 0x0" ::: "memory");
    }
    __syncthreads();

#pragma unroll
    for (int kk = 0; kk < 64; kk += 32) {
      // B fragment: contiguous 32B per lane from pre-swizzled weights
      const _Float16* bp = bbase + ((size_t)((kb + kk) >> 5) << 12);
      const v8h blo = *(const v8h*)bp;
      const v8h bhi = *(const v8h*)(bp + 8);
      const v16h b = __builtin_shufflevector(blo, bhi, 0, 1, 2, 3, 4, 5, 6, 7,
                                             8, 9, 10, 11, 12, 13, 14, 15);
      // A fragment, sub-tile 0 (rows 0..15): row M=ln
      const v8h a0lo = *(const v8h*)(&lds[ln * 72 + kk + (g << 4)]);
      const v8h a0hi = *(const v8h*)(&lds[ln * 72 + kk + 8 + (g << 4)]);
      const v16h a0 = __builtin_shufflevector(a0lo, a0hi, 0, 1, 2, 3, 4, 5, 6,
                                              7, 8, 9, 10, 11, 12, 13, 14, 15);
      c0 = __builtin_amdgcn_wmma_f32_16x16x32_f16(
          false, a0, false, b, (short)0, c0, false, false);
      // A fragment, sub-tile 1 (rows 16..31)
      const v8h a1lo = *(const v8h*)(&lds[(16 + ln) * 72 + kk + (g << 4)]);
      const v8h a1hi =
          *(const v8h*)(&lds[(16 + ln) * 72 + kk + 8 + (g << 4)]);
      const v16h a1 = __builtin_shufflevector(a1lo, a1hi, 0, 1, 2, 3, 4, 5, 6,
                                              7, 8, 9, 10, 11, 12, 13, 14, 15);
      c1 = __builtin_amdgcn_wmma_f32_16x16x32_f16(
          false, a1, false, b, (short)0, c1, false, false);
    }
  }

  // Epilogue: C VGPR r -> row (r + 8*g), col = 16*wv + ln
  const int col = (wv << 4) + ln;
  const float bv  = bias  ? bias[col]  : 0.0f;
  const float b2v = bias2 ? bias2[col] : 0.0f;
#pragma unroll
  for (int r = 0; r < 8; ++r) {
    const int row0 = m0 + r + (g << 3);
    const int row1 = row0 + 16;
    float v0 = c0[r] + bv;
    float v1 = c1[r] + bv;
    if (relu) { v0 = fmaxf(v0, 0.0f); v1 = fmaxf(v1, 0.0f); }
    const size_t o0 = (size_t)row0 * 128 + col;
    const size_t o1 = (size_t)row1 * 128 + col;
    if (out32) { out32[o0] = v0; out32[o1] = v1; }
    if (out16) { out16[o0] = (_Float16)v0; out16[o1] = (_Float16)v1; }
    if (accout) {  // fused GCN self-loop contribution + bias
      const float d0 = dinv[row0], d1 = dinv[row1];
      accout[o0] = d0 * d0 * v0 + b2v;
      accout[o1] = d1 * d1 * v1 + b2v;
    }
  }
}

// ---------------------------------------------------------------------------
// one wave per edge: gather float4/lane from h[src], scatter-atomic to acc[dst]
// ---------------------------------------------------------------------------
__global__ void conv_edge128(const long long* __restrict__ src,
                             const long long* __restrict__ dst,
                             const float* __restrict__ dinv,
                             const float* __restrict__ h,
                             float* __restrict__ acc, int e) {
  int wid = (blockIdx.x * blockDim.x + threadIdx.x) >> 5;
  int lane = threadIdx.x & 31;
  if (wid >= e) return;
  long long s = src[wid], d = dst[wid];
  float w = dinv[s] * dinv[d];
  const float4 v = *(const float4*)(h + ((size_t)s << 7) + (lane << 2));
  float* p = acc + ((size_t)d << 7) + (lane << 2);
  atomicAdd(p + 0, v.x * w);
  atomicAdd(p + 1, v.y * w);
  atomicAdd(p + 2, v.z * w);
  atomicAdd(p + 3, v.w * w);
}

// ---------------------------------------------------------------------------
// Tiny 128->4 GEMM with fused ReLU on the input (fp32, W2 cached in LDS):
// H2[node] = relu(X[node]) @ W2
// ---------------------------------------------------------------------------
__global__ void __launch_bounds__(256)
gemm_x_w2(const float* __restrict__ X, const float* __restrict__ W2,
          float* __restrict__ H2, int n) {
  __shared__ float ws[512];
  int t = threadIdx.x;
  ws[t] = W2[t];
  ws[t + 256] = W2[t + 256];
  __syncthreads();
  int node = blockIdx.x * blockDim.x + t;
  if (node >= n) return;
  float a0 = 0.f, a1 = 0.f, a2 = 0.f, a3 = 0.f;
  const float4* xr = (const float4*)(X + ((size_t)node << 7));
#pragma unroll 4
  for (int k4 = 0; k4 < 32; ++k4) {
    float4 x = xr[k4];
    x.x = fmaxf(x.x, 0.0f); x.y = fmaxf(x.y, 0.0f);
    x.z = fmaxf(x.z, 0.0f); x.w = fmaxf(x.w, 0.0f);
    const float* wp = &ws[k4 * 16];
    a0 += x.x * wp[0] + x.y * wp[4] + x.z * wp[8]  + x.w * wp[12];
    a1 += x.x * wp[1] + x.y * wp[5] + x.z * wp[9]  + x.w * wp[13];
    a2 += x.x * wp[2] + x.y * wp[6] + x.z * wp[10] + x.w * wp[14];
    a3 += x.x * wp[3] + x.y * wp[7] + x.z * wp[11] + x.w * wp[15];
  }
  float4 r = {a0, a1, a2, a3};
  *(float4*)(H2 + ((size_t)node << 2)) = r;
}

__global__ void conv2_self_init(const float* __restrict__ h2,
                                const float* __restrict__ dinv,
                                const float* __restrict__ b2,
                                float* __restrict__ out, int n) {
  int i = blockIdx.x * blockDim.x + threadIdx.x;
  if (i >= n) return;
  float di = dinv[i];
  float s = di * di;
  float4 h = *(const float4*)(h2 + ((size_t)i << 2));
  float4 r = {s * h.x + b2[0], s * h.y + b2[1], s * h.z + b2[2],
              s * h.w + b2[3]};
  *(float4*)(out + ((size_t)i << 2)) = r;
}

__global__ void conv2_edge(const long long* __restrict__ src,
                           const long long* __restrict__ dst,
                           const float* __restrict__ dinv,
                           const float* __restrict__ h2,
                           float* __restrict__ out, int e) {
  int i = blockIdx.x * blockDim.x + threadIdx.x;
  if (i >= e) return;
  long long s = src[i], d = dst[i];
  float w = dinv[s] * dinv[d];
  float4 h = *(const float4*)(h2 + ((size_t)s << 2));
  float* p = out + ((size_t)d << 2);
  atomicAdd(p + 0, h.x * w);
  atomicAdd(p + 1, h.y * w);
  atomicAdd(p + 2, h.z * w);
  atomicAdd(p + 3, h.w * w);
}

// ---------------------------------------------------------------------------
extern "C" void kernel_launch(void* const* d_in, const int* in_sizes, int n_in,
                              void* d_out, int out_size, void* d_ws,
                              size_t ws_size, hipStream_t stream) {
  (void)n_in; (void)out_size; (void)ws_size;
  const float*     clinical = (const float*)d_in[0];
  const float*     mel      = (const float*)d_in[1];
  const long long* ei       = (const long long*)d_in[2];
  const float*     W_mel    = (const float*)d_in[3];
  const float*     b_mel    = (const float*)d_in[4];
  const float*     W_cat    = (const float*)d_in[5];
  const float*     b_cat    = (const float*)d_in[6];
  const float*     W1       = (const float*)d_in[7];
  const float*     b1       = (const float*)d_in[8];
  const float*     W2       = (const float*)d_in[9];
  const float*     b2       = (const float*)d_in[10];
  const int n = in_sizes[0] / 64;
  const int e = in_sizes[2] / 2;
  const long long* esrc = ei;
  const long long* edst = ei + e;

  // scratch carve (256B aligned regions)
  char* wp = (char*)d_ws;
  auto carve = [&](size_t bytes) -> char* {
    char* p = wp;
    wp += (bytes + 255) & ~(size_t)255;
    return p;
  };
  float*     dinv  = (float*)carve((size_t)n * 4);
  _Float16*  WmelH = (_Float16*)carve((size_t)1024 * 128 * 2);
  _Float16*  WcatH = (_Float16*)carve((size_t)192 * 128 * 2);
  _Float16*  W1H   = (_Float16*)carve((size_t)128 * 128 * 2);
  _Float16*  mH    = (_Float16*)carve((size_t)n * 128 * 2);
  _Float16*  xH    = (_Float16*)carve((size_t)n * 128 * 2);
  float*     h1    = (float*)carve((size_t)n * 128 * 4);
  float*     acc1  = (float*)carve((size_t)n * 128 * 4);
  float*     h2    = (float*)carve((size_t)n * 4 * 4);

  // weights -> f16, WMMA fragment order
  swizzle_w_kernel<<<(1024 * 128 + 255) / 256, 256, 0, stream>>>(W_mel, WmelH, 1024, 128);
  swizzle_w_kernel<<<(192 * 128 + 255) / 256, 256, 0, stream>>>(W_cat, WcatH, 192, 128);
  swizzle_w_kernel<<<(128 * 128 + 255) / 256, 256, 0, stream>>>(W1, W1H, 128, 128);

  // symmetric normalization: deg (with self-loop) -> deg^{-1/2}
  fill_f32_kernel<<<(n + 255) / 256, 256, 0, stream>>>(dinv, 1.0f, n);
  deg_count_kernel<<<(e + 255) / 256, 256, 0, stream>>>(edst, dinv, e);
  rsqrt_kernel<<<(n + 255) / 256, 256, 0, stream>>>(dinv, n);

  const int mb = n / 32;  // 100000/32 = 3125, exact
  // m = relu(mel @ W_mel + b_mel), stored f16
  gemm_wmma_kernel<1024, 1024><<<mb, 256, 0, stream>>>(
      mel, 1024, nullptr, 0, WmelH, b_mel, nullptr, mH, /*relu=*/1,
      nullptr, nullptr, nullptr);
  // x = relu([clinical | m] @ W_cat + b_cat), stored f16
  gemm_wmma_kernel<192, 64><<<mb, 256, 0, stream>>>(
      clinical, 64, mH, 128, WcatH, b_cat, nullptr, xH, /*relu=*/1,
      nullptr, nullptr, nullptr);
  // h1 = x @ W1 (f32) with fused conv1 self-loop init:
  // acc1 = dinv^2 * h1 + b1
  gemm_wmma_kernel<128, 0><<<mb, 256, 0, stream>>>(
      nullptr, 0, xH, 128, W1H, nullptr, h1, nullptr, /*relu=*/0,
      acc1, dinv, b1);

  // conv1 edge scatter (ReLU deferred into gemm_x_w2's input read)
  conv_edge128<<<(int)(((size_t)e * 32 + 255) / 256), 256, 0, stream>>>(esrc, edst, dinv, h1, acc1, e);

  // h2 = relu(acc1) @ W2
  gemm_x_w2<<<(n + 255) / 256, 256, 0, stream>>>(acc1, W2, h2, n);

  // conv2 -> d_out
  conv2_self_init<<<(n + 255) / 256, 256, 0, stream>>>(h2, dinv, b2, (float*)d_out, n);
  conv2_edge<<<(e + 255) / 256, 256, 0, stream>>>(esrc, edst, dinv, h2, (float*)d_out, e);
}
